// ModernHopfieldLayer_25151328485833
// MI455X (gfx1250) — matrix operand order repaired and checked by reading, not compiled, not run on hardware
//
#include <hip/hip_runtime.h>
#include <hip/hip_bf16.h>

// ModernHopfieldLayer for MI455X (gfx1250, wave32, WMMA).
// All GEMMs via v_wmma_f32_16x16x32_bf16 with fp32 -> bf16 hi/lo split
// (3-term Markidis) for near-fp32 accuracy at bf16 matrix-core rate.
// All operand conversion is hoisted out of the hot loops into one-pass
// elementwise kernels so the WMMA loops are load+WMMA only.

typedef __bf16 bf16_t;
typedef bf16_t bf16x16 __attribute__((ext_vector_type(16)));
typedef bf16_t bf16x8  __attribute__((ext_vector_type(8)));
typedef bf16_t bf16x4  __attribute__((ext_vector_type(4)));
typedef float  f32x8   __attribute__((ext_vector_type(8)));

#define BETA_F   16.0f
#define LOG2E_F  1.44269504088896340736f
#define NHEADS   16
#define HD       64
#define SEQ_L    2048
#define MEM_M    2048
#define DIM      1024
#define NROWS    4096   /* B * L */

static __device__ __forceinline__ f32x8
wmma_bf16(bf16x16 a, bf16x16 b, f32x8 c) {
  return __builtin_amdgcn_wmma_f32_16x16x32_bf16(false, a, false, b,
                                                 (short)0, c, false, false);
}

// 3-term split product: A*B ~= Ah*Bh + Ah*Bl + Al*Bh (lo*lo ~2^-32, dropped)
static __device__ __forceinline__ f32x8
wmma3(bf16x16 ah, bf16x16 al, bf16x16 bh, bf16x16 bl, f32x8 c) {
  c = wmma_bf16(ah, bh, c);
  c = wmma_bf16(ah, bl, c);
  c = wmma_bf16(al, bh, c);
  return c;
}

static __device__ __forceinline__ void split1(float v, bf16_t& h, bf16_t& l) {
  h = (bf16_t)v;
  l = (bf16_t)(v - (float)h);
}

// A-operand (16x32, 16-bit) from a pre-split bf16 row:
//   elements 0..7  -> K = 8*half + 0..7
//   elements 8..15 -> K = 8*half + 16..23
static __device__ __forceinline__ bf16x16
load_a_bf16(const bf16_t* rowp, int half) {
  const bf16_t* p = rowp + 8 * half;
  bf16x8 a = *(const bf16x8*)(p);
  bf16x8 b = *(const bf16x8*)(p + 16);
  bf16x16 r;
#pragma unroll
  for (int i = 0; i < 8; ++i) { r[i] = a[i]; r[8 + i] = b[i]; }
  return r;
}

// B-operand (32x16, 16-bit): elements 0..7 -> K = 8*half + 0..7, 8..15 -> K = 8*half + 16..23.
static __device__ __forceinline__ bf16x16
load_b_bf16(const bf16_t* rowp, int half) {
  const bf16_t* p = rowp + 8 * half;
  bf16x8 a = *(const bf16x8*)(p);
  bf16x8 b = *(const bf16x8*)(p + 16);
  bf16x16 r;
#pragma unroll
  for (int i = 0; i < 8; ++i) { r[i] = a[i]; r[8 + i] = b[i]; }
  return r;
}

// A-operand from fp32 row with on-the-fly split (used only for fresh data: Q, P)
static __device__ __forceinline__ void
load_a_split(const float* rowp, int half, bf16x16& hi, bf16x16& lo) {
  const float* p = rowp + 8 * half;
#pragma unroll
  for (int i = 0; i < 8; ++i) {
    bf16_t h, l; split1(p[i], h, l);
    hi[i] = h; lo[i] = l;
  }
#pragma unroll
  for (int i = 0; i < 8; ++i) {
    bf16_t h, l; split1(p[16 + i], h, l);
    hi[8 + i] = h; lo[8 + i] = l;
  }
}

// -------------------------------------------------------------------------
// One-pass fp32 -> bf16 hi/lo split (vectorized by 4). n4 = n/4.
// -------------------------------------------------------------------------
__global__ __launch_bounds__(256)
void convert_split_f32(const float* __restrict__ src, bf16_t* __restrict__ h,
                       bf16_t* __restrict__ l, int n4) {
  const int i = blockIdx.x * 256 + threadIdx.x;
  if (i >= n4) return;
  const float4 v = ((const float4*)src)[i];
  bf16x4 hv, lv;
  bf16_t th, tl;
  split1(v.x, th, tl); hv[0] = th; lv[0] = tl;
  split1(v.y, th, tl); hv[1] = th; lv[1] = tl;
  split1(v.z, th, tl); hv[2] = th; lv[2] = tl;
  split1(v.w, th, tl); hv[3] = th; lv[3] = tl;
  ((bf16x4*)h)[i] = hv;
  ((bf16x4*)l)[i] = lv;
}

// -------------------------------------------------------------------------
// Memory bank [M,64] -> bf16 hi/lo in row-major AND transposed [64,M] layouts.
// -------------------------------------------------------------------------
__global__ __launch_bounds__(256)
void convert_mem_split(const float* __restrict__ mem,
                       bf16_t* __restrict__ mh, bf16_t* __restrict__ ml,
                       bf16_t* __restrict__ mth, bf16_t* __restrict__ mtl) {
  const int idx = blockIdx.x * 256 + threadIdx.x;   // M*HD threads
  const int m = idx >> 6;
  const int k = idx & 63;
  bf16_t h, l;
  split1(mem[idx], h, l);
  mh[idx] = h;
  ml[idx] = l;
  mth[(size_t)k * MEM_M + m] = h;
  mtl[(size_t)k * MEM_M + m] = l;
}

// -------------------------------------------------------------------------
// Y[4096,1024] = X @ W^T + bias from pre-split bf16 operands.
// One wave -> one 16x16 tile. grid = (N/16, M/128), block = 256 (8 waves).
// -------------------------------------------------------------------------
__global__ __launch_bounds__(256)
void gemm_bias_wmma(const bf16_t* __restrict__ Xh, const bf16_t* __restrict__ Xl,
                    const bf16_t* __restrict__ Wh, const bf16_t* __restrict__ Wl,
                    const float* __restrict__ bias, float* __restrict__ Y) {
  const int lane = threadIdx.x & 31;
  const int wave = threadIdx.x >> 5;
  const int half = lane >> 4;
  const int ln   = lane & 15;
  const int row0 = blockIdx.y * 128 + wave * 16;
  const int col0 = blockIdx.x * 16;

  const size_t aoff = (size_t)(row0 + ln) * DIM;   // A row for this lane
  const size_t boff = (size_t)(col0 + ln) * DIM;   // W row n = B column n

  f32x8 acc = {};
  for (int k0 = 0; k0 < DIM; k0 += 32) {
    bf16x16 ah = load_a_bf16(Xh + aoff + k0, half);
    bf16x16 al = load_a_bf16(Xl + aoff + k0, half);
    bf16x16 bh = load_b_bf16(Wh + boff + k0, half);
    bf16x16 bl = load_b_bf16(Wl + boff + k0, half);
    acc = wmma3(ah, al, bh, bl, acc);
  }

  const float bv = bias[col0 + ln];
  const int rbase = half * 8;   // C layout: VGPR r -> row r + 8*half
#pragma unroll
  for (int r = 0; r < 8; ++r)
    Y[(size_t)(row0 + rbase + r) * DIM + (col0 + ln)] = acc[r] + bv;
}

// -------------------------------------------------------------------------
// One Hopfield retrieval iteration, flash-attention style, in-place on qbuf.
// qbuf layout: [B, L, D], head h occupies cols h*64..h*64+63.
// grid = (L/128, B*H); block = 256; wave w owns q-rows [bx*128+16w, +16).
// Memory-bank B operands come pre-split/pre-transposed from L2-resident
// global arrays, so there is no block-level synchronization at all.
// -------------------------------------------------------------------------
__global__ __launch_bounds__(256)
void hopfield_flash_wmma(float* __restrict__ qbuf,
                         const bf16_t* __restrict__ mh, const bf16_t* __restrict__ ml,
                         const bf16_t* __restrict__ mth, const bf16_t* __restrict__ mtl) {
  __shared__ alignas(16) float s_p[8][16 * 32];   // per-wave P transpose staging

  const int lane = threadIdx.x & 31;
  const int wave = threadIdx.x >> 5;
  const int half = lane >> 4;
  const int ln   = lane & 15;
  const int b    = blockIdx.y >> 4;
  const int h    = blockIdx.y & (NHEADS - 1);
  const int row0 = blockIdx.x * 128 + wave * 16;

  // This wave's Q block (16 rows x 64) as two A-fragments, hi/lo split once.
  const float* qrow = qbuf + ((size_t)(b * SEQ_L + row0 + ln)) * DIM + h * HD;
  bf16x16 qh[2], ql[2];
#pragma unroll
  for (int kc = 0; kc < 2; ++kc)
    load_a_split(qrow + kc * 32, half, qh[kc], ql[kc]);

  f32x8 o[4] = {};          // O accumulator: 16 rows x 64 cols
  float mi[8], li[8];
#pragma unroll
  for (int r = 0; r < 8; ++r) { mi[r] = -1e30f; li[r] = 0.0f; }

  for (int m0 = 0; m0 < MEM_M; m0 += 32) {
    if (m0 + 32 < MEM_M)   // hint next tile toward the caches
      __builtin_prefetch(mh + (size_t)(m0 + 32 + ln) * HD, 0, 0);

    // S = Q @ mem_tile^T : two 16x16 fragments (m-cols nt*16..nt*16+15)
    f32x8 sfrag[2];
#pragma unroll
    for (int nt = 0; nt < 2; ++nt) {
      f32x8 c = {};
      const size_t mrow = (size_t)(m0 + nt * 16 + ln) * HD;
#pragma unroll
      for (int kc = 0; kc < 2; ++kc) {
        bf16x16 bh = load_b_bf16(mh + mrow + kc * 32, half);
        bf16x16 bl = load_b_bf16(ml + mrow + kc * 32, half);
        c = wmma3(qh[kc], ql[kc], bh, bl, c);
      }
      sfrag[nt] = c;
    }
#pragma unroll
    for (int nt = 0; nt < 2; ++nt)
#pragma unroll
      for (int r = 0; r < 8; ++r) sfrag[nt][r] *= BETA_F;   // logits

    // ---- online softmax (each row lives across 16 lanes of a half-wave) ----
    float mnew[8], corr[8];
#pragma unroll
    for (int r = 0; r < 8; ++r) {
      float lm = fmaxf(sfrag[0][r], sfrag[1][r]);
      lm = fmaxf(lm, __shfl_xor(lm, 1, 32));
      lm = fmaxf(lm, __shfl_xor(lm, 2, 32));
      lm = fmaxf(lm, __shfl_xor(lm, 4, 32));
      lm = fmaxf(lm, __shfl_xor(lm, 8, 32));
      mnew[r] = fmaxf(mi[r], lm);
      corr[r] = __builtin_amdgcn_exp2f((mi[r] - mnew[r]) * LOG2E_F);
    }
#pragma unroll
    for (int nt = 0; nt < 2; ++nt)
#pragma unroll
      for (int r = 0; r < 8; ++r)
        sfrag[nt][r] = __builtin_amdgcn_exp2f((sfrag[nt][r] - mnew[r]) * LOG2E_F);
#pragma unroll
    for (int r = 0; r < 8; ++r) {
      float s = sfrag[0][r] + sfrag[1][r];
      s += __shfl_xor(s, 1, 32);
      s += __shfl_xor(s, 2, 32);
      s += __shfl_xor(s, 4, 32);
      s += __shfl_xor(s, 8, 32);
      li[r] = li[r] * corr[r] + s;
      mi[r] = mnew[r];
    }
#pragma unroll
    for (int t = 0; t < 4; ++t)
#pragma unroll
      for (int r = 0; r < 8; ++r) o[t][r] *= corr[r];

    // ---- transpose P (C layout -> A layout) via per-wave LDS staging ----
    float* pst = &s_p[wave][0];
#pragma unroll
    for (int nt = 0; nt < 2; ++nt)
#pragma unroll
      for (int r = 0; r < 8; ++r)
        pst[(r + half * 8) * 32 + nt * 16 + ln] = sfrag[nt][r];
    // wave-private region + wave32 lockstep: DS-count drain is sufficient
    asm volatile("s_wait_dscnt 0" ::: "memory");

    bf16x16 ph, pl;
    load_a_split(&pst[ln * 32], half, ph, pl);

    // O += P @ mem_tile  (B from pre-transposed global arrays; K = 32 m-rows)
#pragma unroll
    for (int t = 0; t < 4; ++t) {
      const size_t trow = (size_t)(t * 16 + ln) * MEM_M + m0;
      bf16x16 bh = load_b_bf16(mth + trow, half);
      bf16x16 bl = load_b_bf16(mtl + trow, half);
      o[t] = wmma3(ph, pl, bh, bl, o[t]);
    }
  }

  // normalize rows and write retrieved q back in place
  float* orow = qbuf + ((size_t)(b * SEQ_L + row0)) * DIM + h * HD;
#pragma unroll
  for (int r = 0; r < 8; ++r) {
    const float inv = 1.0f / li[r];
#pragma unroll
    for (int t = 0; t < 4; ++t)
      orow[(size_t)(r + half * 8) * DIM + t * 16 + ln] = o[t][r] * inv;
  }
}

// -------------------------------------------------------------------------
extern "C" void kernel_launch(void* const* d_in, const int* in_sizes, int n_in,
                              void* d_out, int out_size, void* d_ws, size_t ws_size,
                              hipStream_t stream) {
  const float* x      = (const float*)d_in[0];
  const float* memory = (const float*)d_in[1];
  const float* Wq     = (const float*)d_in[2];
  const float* bq     = (const float*)d_in[3];
  const float* Wo     = (const float*)d_in[4];
  const float* bo     = (const float*)d_in[5];
  // d_in[6] = iterations (fixed at 2 by the reference setup)
  float* out = (float*)d_out;

  // workspace layout (256B-aligned offsets)
  char* ws = (char*)d_ws;
  float*  qbuf = (float*)(ws);                       // 16 MB
  bf16_t* ah   = (bf16_t*)(ws + (16u << 20));        //  8 MB (X / qbuf hi)
  bf16_t* al   = (bf16_t*)(ws + (24u << 20));        //  8 MB (X / qbuf lo)
  bf16_t* wqh  = (bf16_t*)(ws + (32u << 20));        //  2 MB
  bf16_t* wql  = (bf16_t*)(ws + (34u << 20));        //  2 MB
  bf16_t* woh  = (bf16_t*)(ws + (36u << 20));        //  2 MB
  bf16_t* wol  = (bf16_t*)(ws + (38u << 20));        //  2 MB
  bf16_t* mh   = (bf16_t*)(ws + (40u << 20));        // 256 KB
  bf16_t* ml   = (bf16_t*)(ws + (40u << 20) + (256u << 10));
  bf16_t* mth  = (bf16_t*)(ws + (40u << 20) + (512u << 10));
  bf16_t* mtl  = (bf16_t*)(ws + (40u << 20) + (768u << 10));

  dim3 blk(256);
  const int nX4 = NROWS * DIM / 4;   // 1M float4
  const int nW4 = DIM * DIM / 4;     // 256K float4

  // one-pass operand pre-splits (bandwidth-trivial, removes all loop VALU)
  convert_split_f32<<<dim3((nX4 + 255) / 256), blk, 0, stream>>>(x, ah, al, nX4);
  convert_split_f32<<<dim3((nW4 + 255) / 256), blk, 0, stream>>>(Wq, wqh, wql, nW4);
  convert_split_f32<<<dim3((nW4 + 255) / 256), blk, 0, stream>>>(Wo, woh, wol, nW4);
  convert_mem_split<<<dim3(MEM_M * HD / 256), blk, 0, stream>>>(memory, mh, ml, mth, mtl);

  // q = x @ Wq^T + bq
  gemm_bias_wmma<<<dim3(DIM / 16, NROWS / 128), blk, 0, stream>>>(ah, al, wqh, wql, bq, qbuf);

  // two Hopfield retrieval iterations (in-place on qbuf)
  hopfield_flash_wmma<<<dim3(SEQ_L / 128, 2 * NHEADS), blk, 0, stream>>>(qbuf, mh, ml, mth, mtl);
  hopfield_flash_wmma<<<dim3(SEQ_L / 128, 2 * NHEADS), blk, 0, stream>>>(qbuf, mh, ml, mth, mtl);

  // out = q @ Wo^T + bo  (re-split retrieved q first)
  convert_split_f32<<<dim3((nX4 + 255) / 256), blk, 0, stream>>>(qbuf, ah, al, nX4);
  gemm_bias_wmma<<<dim3(DIM / 16, NROWS / 128), blk, 0, stream>>>(ah, al, woh, wol, bo, out);
}
